// CPCUnsupersivedCriterion_42606075576641
// MI455X (gfx1250) — compile-verified, matrix operand order checked by reading, not run on hardware
//
#include <hip/hip_runtime.h>
#include <hip/hip_bf16.h>

typedef __attribute__((ext_vector_type(16))) _Float16 v16h;
typedef __attribute__((ext_vector_type(8)))  float    v8f;
typedef __attribute__((ext_vector_type(4)))  float    v4f;

#define B_    8
#define S_    128
#define KPRED 12
#define DAR   256
#define DENC  256
#define NNEG  128
#define W_    116                 // S - K
#define MDIM  (B_ * W_)           // 928
#define NDIM  (KPRED * DENC)      // 3072

// Load one A/B fragment for v_wmma_f32_16x16x32_f16 from an f32 row.
// Per the CDNA5 ISA 16-bit A layout: lanes 0-15 hold K in [0,8)+[16,24),
// lanes 16-31 hold K in [8,16)+[24,32).  'koff' = kk + (lane<16 ? 0 : 8).
__device__ inline v16h load_frag_f32(const float* __restrict__ row, int koff) {
    v4f a0 = *(const v4f*)(row + koff);
    v4f a1 = *(const v4f*)(row + koff + 4);
    v4f b0 = *(const v4f*)(row + koff + 16);
    v4f b1 = *(const v4f*)(row + koff + 20);
    v16h h;
#pragma unroll
    for (int i = 0; i < 4; ++i) {
        h[i]      = (_Float16)a0[i];
        h[4 + i]  = (_Float16)a1[i];
        h[8 + i]  = (_Float16)b0[i];
        h[12 + i] = (_Float16)b1[i];
    }
    return h;
}

__global__ void zero_out_kernel(float* __restrict__ out) {
    if (threadIdx.x < 2 * KPRED) out[threadIdx.x] = 0.0f;
}

// Stage 1: locC_flat(928,3072) = c(928,256) x WpredFlat^T(256,3072)
// Output layout: locC[bw][k*256 + e]  (each bw's block is the stage-2 A tile).
__global__ __launch_bounds__(256) void locc_gemm_kernel(
    const float* __restrict__ cFeature,   // (B,S,DAR)
    const float* __restrict__ Wpred,      // (K,DENC,DAR) == flat (3072,256)
    float* __restrict__ locC)             // (928,3072) + pad
{
    const int lane = threadIdx.x & 31;
    const int wave = threadIdx.x >> 5;
    const int tile = blockIdx.x * 8 + wave;     // 58*192 = 11136 tiles
    const int mt = tile / 192;
    const int nt = tile - mt * 192;
    const int hl = lane & 15;

    // A row: m = b*W + w  ->  cFeature[b, w, :]
    const int m = mt * 16 + hl;
    const int b = m / W_;
    const int w = m - b * W_;
    const float* arow = cFeature + (size_t)(b * S_ + w) * DAR;
    // B^T row n = WpredFlat[n, :] (contiguous over d)
    const int n = nt * 16 + hl;
    const float* brow = Wpred + (size_t)n * DAR;

    const int kbase = (lane < 16) ? 0 : 8;
    v8f acc = {};
#pragma unroll
    for (int kk = 0; kk < DAR; kk += 32) {
        v16h a  = load_frag_f32(arow, kk + kbase);
        v16h bf = load_frag_f32(brow, kk + kbase);
        acc = __builtin_amdgcn_wmma_f32_16x16x32_f16(
            false, a, false, bf, (short)0, acc, false, false);
    }
    // D layout: VGPR j holds (m = mt*16 + j + (lane<16?0:8), n = nt*16 + lane%16)
    const int mrow0 = mt * 16 + kbase;
    const int ncol  = nt * 16 + hl;
#pragma unroll
    for (int j = 0; j < 8; ++j)
        locC[(size_t)(mrow0 + j) * NDIM + ncol] = acc[j];
}

// Stage 2: per (b,w) block: negScore (8 waves x 16 negatives via WMMA),
// posScore (wave 0 extra tile, diagonal), then LSE/argmax reduction.
__global__ __launch_bounds__(256) void score_reduce_kernel(
    const float* __restrict__ locC,       // (928,3072) + pad
    const float* __restrict__ enc,        // (B,S,DENC)
    const int*   __restrict__ batchIdx,   // (B*NNEG*W)
    const int*   __restrict__ seqIdx,     // (B*NNEG*W)
    float* __restrict__ out)              // 24 floats: losses[12], accs[12]
{
    __shared__ float sc[16][132];         // rows k, col0 = pos, cols 1..128 = negs

    const int bw   = blockIdx.x;          // 0..927
    const int b    = bw / W_;
    const int w    = bw - b * W_;
    const int lane = threadIdx.x & 31;
    const int wave = threadIdx.x >> 5;
    const int hl   = lane & 15;
    const int kbase = (lane < 16) ? 0 : 8;

    // A tile row (k = hl) of this (b,w); rows 12..15 read pad (ignored).
    const float* arow = locC + (size_t)bw * NDIM + (size_t)hl * DENC;

    // Gathered negative row for this lane's column.
    const int nn = wave * 16 + hl;                         // negative index 0..127
    const int i  = (b * NNEG + nn) * W_ + w;
    const int r  = (seqIdx[i] + w) % S_ + batchIdx[i] * S_;
    const float* brow = enc + (size_t)r * DENC;

    v8f accn = {};
#pragma unroll
    for (int kk = 0; kk < DENC; kk += 32) {
        v16h a  = load_frag_f32(arow, kk + kbase);
        v16h bf = load_frag_f32(brow, kk + kbase);
        accn = __builtin_amdgcn_wmma_f32_16x16x32_f16(
            false, a, false, bf, (short)0, accn, false, false);
    }
    const float sinv = 1.0f / (float)DENC;
#pragma unroll
    for (int j = 0; j < 8; ++j)
        sc[kbase + j][1 + wave * 16 + hl] = accn[j] * sinv;

    if (wave == 0) {
        // Positive tile: column n -> enc[b, w+n+1] (clamped; n>=12 unused)
        int rp = w + hl + 1;
        if (rp > S_ - 1) rp = S_ - 1;
        const float* prow = enc + ((size_t)b * S_ + rp) * DENC;
        v8f accp = {};
#pragma unroll
        for (int kk = 0; kk < DENC; kk += 32) {
            v16h a  = load_frag_f32(arow, kk + kbase);
            v16h bf = load_frag_f32(prow, kk + kbase);
            accp = __builtin_amdgcn_wmma_f32_16x16x32_f16(
                false, a, false, bf, (short)0, accp, false, false);
        }
        float cp[8];
#pragma unroll
        for (int j = 0; j < 8; ++j) cp[j] = accp[j] * sinv;
        // Extract diagonal D[k][k], k = 0..11
#pragma unroll
        for (int k = 0; k < KPRED; ++k) {
            const int srcreg  = (k < 8) ? k : (k - 8);
            const int srclane = (k < 8) ? k : (k + 16);
            float v = __shfl(cp[srcreg], srclane, 32);
            if (lane == 0) sc[k][0] = v;
        }
    }
    __syncthreads();

    if (wave == 0 && lane < KPRED) {
        const int k = lane;
        const float pos = sc[k][0];
        float mx = pos, mneg = -3.0e38f;
        for (int j = 1; j <= NNEG; ++j) {
            const float v = sc[k][j];
            mx   = fmaxf(mx, v);
            mneg = fmaxf(mneg, v);
        }
        float s = 0.0f;
        for (int j = 0; j <= NNEG; ++j) s += __expf(sc[k][j] - mx);
        const float lse = mx + __logf(s);
        const float inv = 1.0f / (float)MDIM;
        atomicAdd(&out[k],         (lse - pos) * inv);
        atomicAdd(&out[KPRED + k], (pos >= mneg) ? inv : 0.0f);
    }
}

extern "C" void kernel_launch(void* const* d_in, const int* in_sizes, int n_in,
                              void* d_out, int out_size, void* d_ws, size_t ws_size,
                              hipStream_t stream) {
    const float* cFeature = (const float*)d_in[0];  // (B,S,DAR)
    const float* enc      = (const float*)d_in[1];  // (B,S,DENC)
    const float* Wpred    = (const float*)d_in[2];  // (K,DENC,DAR)
    const int*   batchIdx = (const int*)d_in[3];
    const int*   seqIdx   = (const int*)d_in[4];
    float*       out      = (float*)d_out;          // 24 floats
    float*       locC     = (float*)d_ws;           // 928*3072 f32 + 4KB pad

    zero_out_kernel<<<1, 32, 0, stream>>>(out);

    // 58*192 tiles, 8 waves (256 threads) per block
    locc_gemm_kernel<<<(58 * 192) / 8, 256, 0, stream>>>(cFeature, Wpred, locC);

    score_reduce_kernel<<<MDIM, 256, 0, stream>>>(locC, enc, batchIdx, seqIdx, out);
}
